// PointNetRes_60696477827768
// MI455X (gfx1250) — matrix operand order repaired and checked by reading, not compiled, not run on hardware
//
#include <hip/hip_runtime.h>
#include <math.h>

// ---------------------------------------------------------------------------
// MI455X (gfx1250) implementation of the PointNetRes reference.
//  * All 1x1 convs run as bf16 WMMA GEMMs (v_wmma_f32_16x16x32_bf16),
//    fp32 accumulate. Each wave register-blocks a 32x32 output tile
//    (2x2 WMMA accumulators) -> 2 LDS b128 loads per WMMA instead of 4.
//  * Training-mode BN handled two-pass: conv kernels emit raw (pre-BN)
//    bf16 activations + per-channel sum/sumsq partials (wave-shuffle
//    reduced, then global atomicAdd); a finalize kernel folds stats into
//    scale/shift which the *consumer* conv applies while staging its LDS
//    input tile (fused BN+ReLU on the load path).
//  * SoftPool: per (b,k) top-16-min selection via LDS-resident values and
//    u64 (value,index) keyed min-reductions, then a small gather-dot.
// ---------------------------------------------------------------------------

typedef __attribute__((ext_vector_type(16))) __bf16 bf16x16;
typedef __attribute__((ext_vector_type(8)))  float  f32x8;

union FragBF { bf16x16 v; uint4 q[2]; };

__device__ __forceinline__ unsigned short f2bf_bits(float f) {
  unsigned u = __float_as_uint(f);
  unsigned r = u + 0x7FFFu + ((u >> 16) & 1u);   // round-to-nearest-even
  return (unsigned short)(r >> 16);
}
__device__ __forceinline__ float bf2f_bits(unsigned short h) {
  return __uint_as_float(((unsigned)h) << 16);
}

// ---------------------------------------------------------------------------
// x (fp32) -> bf16 raw buffer
// ---------------------------------------------------------------------------
__global__ void to_bf16_kernel(const float* __restrict__ x,
                               unsigned short* __restrict__ o, int n) {
  int i = blockIdx.x * blockDim.x + threadIdx.x;
  if (i < n) o[i] = f2bf_bits(x[i]);
}

// ---------------------------------------------------------------------------
// Generic fused 1x1-conv GEMM:
//   Y[b, m, n] = sum_c W[m,c] * X[b, c, n] + bias[m]
// X assembled on the fly: optional 32 "glob" constant channels, then raw
// bf16 activations with optional scale/shift (producer BN) and optional
// ReLU. Output: raw bf16 (+ per-channel sum/sumsq stats) or fp32 tanh head.
//
// Block: 256 threads = 8 waves as 2(M)x4(N); each wave computes a 32x32
// macro-tile via 2x2 WMMA accumulators (A/B fragments reused across the
// register block). Block tile: 64 outputs x 128 points.
// ---------------------------------------------------------------------------
template<int GLOBCH, bool IN_BN, bool IN_RELU, bool OUT_STATS, bool OUT_TANH>
__global__ __launch_bounds__(256)
void conv1x1_wmma(const unsigned short* __restrict__ in_raw, // [B,CinRaw,N] bf16
                  const float* __restrict__ glob,            // [B,GLOBCH]
                  const float* __restrict__ in_ss,           // scale[0..CinRaw), shift[CinRaw..)
                  const float* __restrict__ W,               // [Cout,Cin] fp32
                  const float* __restrict__ bias,            // [Cout]
                  unsigned short* __restrict__ out_raw,      // [B,Cout,N] bf16
                  float* __restrict__ out_f32,               // tanh head output
                  float* __restrict__ stats,                 // [2*Cout]
                  int Cin, int Cout, int N) {
  const int tid   = threadIdx.x;
  const int nBase = blockIdx.x * 128;
  const int mBase = blockIdx.y * 64;
  const int b     = blockIdx.z;
  const int KP    = (Cin + 31) & ~31;     // K padded to WMMA depth
  const int LDSTR = KP + 8;               // +8 halfwords: bank skew, keeps 16B align
  const int CinRaw = Cin - GLOBCH;

  extern __shared__ unsigned short smem_us[];
  unsigned short* wl = smem_us;                    // [64][LDSTR]  weights
  unsigned short* xl = smem_us + 64 * LDSTR;       // [128][LDSTR] activations (n-major)

  // ---- stage weight tile (zero-padded in m and k) ----
  for (int i = tid; i < 64 * KP; i += 256) {
    int r = i / KP, c = i - r * KP;
    int m = mBase + r;
    float v = (m < Cout && c < Cin) ? W[(size_t)m * Cin + c] : 0.0f;
    wl[r * LDSTR + c] = f2bf_bits(v);
  }

  // ---- stage activation tile, fusing BN(scale/shift)+ReLU of the producer ----
  // i = c*128 + n : consecutive threads walk consecutive n -> coalesced reads
  for (int i = tid; i < 128 * KP; i += 256) {
    int n = i & 127, c = i >> 7;
    float v = 0.0f;
    if (c < Cin) {
      if (GLOBCH > 0 && c < GLOBCH) {
        v = glob[b * GLOBCH + c];
      } else {
        int cr = c - GLOBCH;
        v = bf2f_bits(in_raw[((size_t)b * CinRaw + cr) * N + nBase + n]);
        if (IN_BN)   v = v * in_ss[cr] + in_ss[CinRaw + cr];
        if (IN_RELU) v = fmaxf(v, 0.0f);
      }
    }
    xl[n * LDSTR + c] = f2bf_bits(v);
  }
  __syncthreads();

  // ---- WMMA K-loop, 2x2 register block per wave ----
  const int wv = tid >> 5, lane = tid & 31;
  const int wm = wv & 1, wn = wv >> 1;        // wave macro-tile: (wm*32) x (wn*32)
  const int half = lane >> 4, l16 = lane & 15;
  const unsigned short* arow0 = wl + (wm * 32 + l16) * LDSTR;       // M sub 0
  const unsigned short* arow1 = arow0 + 16 * LDSTR;                 // M sub 1
  const unsigned short* brow0 = xl + (wn * 32 + l16) * LDSTR;       // N sub 0
  const unsigned short* brow1 = brow0 + 16 * LDSTR;                 // N sub 1

  f32x8 acc00 = {}, acc01 = {}, acc10 = {}, acc11 = {};
  for (int k0 = 0; k0 < KP; k0 += 32) {
    FragBF a0, a1, b0, b1;
    // A 16x32 bf16 layout: per half-wave, K groups of 8 with a +16 jump
    a0.q[0] = *reinterpret_cast<const uint4*>(arow0 + k0 + 8 * half);
    a0.q[1] = *reinterpret_cast<const uint4*>(arow0 + k0 + 16 + 8 * half);
    a1.q[0] = *reinterpret_cast<const uint4*>(arow1 + k0 + 8 * half);
    a1.q[1] = *reinterpret_cast<const uint4*>(arow1 + k0 + 16 + 8 * half);
    // B 32x16 bf16: same K order as A (groups of 8 with a +16 jump)
    b0.q[0] = *reinterpret_cast<const uint4*>(brow0 + k0 + 8 * half);
    b0.q[1] = *reinterpret_cast<const uint4*>(brow0 + k0 + 16 + 8 * half);
    b1.q[0] = *reinterpret_cast<const uint4*>(brow1 + k0 + 8 * half);
    b1.q[1] = *reinterpret_cast<const uint4*>(brow1 + k0 + 16 + 8 * half);
    acc00 = __builtin_amdgcn_wmma_f32_16x16x32_bf16(false, a0.v, false, b0.v,
                                                    (short)0, acc00, false, false);
    acc01 = __builtin_amdgcn_wmma_f32_16x16x32_bf16(false, a0.v, false, b1.v,
                                                    (short)0, acc01, false, false);
    acc10 = __builtin_amdgcn_wmma_f32_16x16x32_bf16(false, a1.v, false, b0.v,
                                                    (short)0, acc10, false, false);
    acc11 = __builtin_amdgcn_wmma_f32_16x16x32_bf16(false, a1.v, false, b1.v,
                                                    (short)0, acc11, false, false);
  }

  // ---- epilogue: bias, store raw bf16 / tanh fp32, accumulate BN stats ----
  f32x8 accs[2][2] = {{acc00, acc01}, {acc10, acc11}};
#pragma unroll
  for (int sm = 0; sm < 2; ++sm) {
#pragma unroll
    for (int sn = 0; sn < 2; ++sn) {
      const int nCol = nBase + wn * 32 + sn * 16 + l16;
      float ysum[8], ysq[8];
#pragma unroll
      for (int e = 0; e < 8; ++e) {
        int m = mBase + wm * 32 + sm * 16 + 8 * half + e; // C/D: VGPR e -> M=e+8*half
        float y = accs[sm][sn][e] + ((m < Cout) ? bias[m] : 0.0f);
        if (m < Cout) {
          if (OUT_TANH) out_f32[((size_t)b * Cout + m) * N + nCol] = tanhf(y);
          else          out_raw[((size_t)b * Cout + m) * N + nCol] = f2bf_bits(y);
        }
        ysum[e] = (m < Cout) ? y : 0.0f;
        ysq[e]  = (m < Cout) ? y * y : 0.0f;
      }
      if (OUT_STATS) {
#pragma unroll
        for (int e = 0; e < 8; ++e) {
          float s = ysum[e], q = ysq[e];
          // reduce over the 16 lanes of this half-wave (xor masks stay in half)
          for (int off = 1; off < 16; off <<= 1) {
            s += __shfl_xor(s, off, 32);
            q += __shfl_xor(q, off, 32);
          }
          if (l16 == 0) {
            int m = mBase + wm * 32 + sm * 16 + 8 * half + e;
            if (m < Cout) {
              atomicAdd(&stats[m], s);
              atomicAdd(&stats[Cout + m], q);
            }
          }
        }
      }
    }
  }
}

// ---------------------------------------------------------------------------
// BN finalize: stats (sum,sumsq) -> per-channel scale/shift
// ---------------------------------------------------------------------------
__global__ void bn_finalize(const float* __restrict__ stats,
                            const float* __restrict__ g,
                            const float* __restrict__ be,
                            float* __restrict__ ss, int C, float invCnt) {
  int c = blockIdx.x * blockDim.x + threadIdx.x;
  if (c >= C) return;
  float m   = stats[c] * invCnt;
  float var = stats[C + c] * invCnt - m * m;
  float sc  = g[c] * rsqrtf(var + 1e-5f);
  ss[c]     = sc;
  ss[C + c] = be[c] - m * sc;
}

// ---------------------------------------------------------------------------
// SoftPool select: per (b,k) the indices of the 16 smallest h3[b,k,:]
// (stable tie-break by index, matching argsort). One block per (b,k);
// values staged once in LDS as bf16, 16 rounds of u64-keyed min reduction.
// ---------------------------------------------------------------------------
__global__ __launch_bounds__(256)
void softpool_select(const unsigned short* __restrict__ raw3,
                     const float* __restrict__ ss3,
                     int* __restrict__ idx, int N) {
  __shared__ unsigned short sv[16384];
  __shared__ unsigned long long red[256];
  const int bk = blockIdx.x, b = bk >> 5, k = bk & 31, tid = threadIdx.x;
  const float sc = ss3[k], sh = ss3[32 + k];
  const unsigned short* row = raw3 + ((size_t)b * 32 + k) * N;
  for (int n = tid; n < N; n += 256)
    sv[n] = f2bf_bits(bf2f_bits(row[n]) * sc + sh);
  __syncthreads();

  unsigned long long lastKey = 0;   // keys are >= 1 and strictly increasing
  for (int p = 0; p < 16; ++p) {
    unsigned long long best = ~0ull;
    for (int n = tid; n < N; n += 256) {
      unsigned f = ((unsigned)sv[n]) << 16;
      unsigned ord = (f & 0x80000000u) ? ~f : (f | 0x80000000u); // ascending map
      unsigned long long key = (((unsigned long long)ord + 1ull) << 14) | (unsigned)n;
      if (key > lastKey && key < best) best = key;
    }
    red[tid] = best;
    __syncthreads();
    for (int s = 128; s > 0; s >>= 1) {
      if (tid < s && red[tid + s] < red[tid]) red[tid] = red[tid + s];
      __syncthreads();
    }
    best = red[0];
    lastKey = best;
    if (tid == 0) idx[bk * 16 + p] = (int)(best & 0x3FFFu);
    __syncthreads();
  }
}

// ---------------------------------------------------------------------------
// SoftPool gather + conv9: v[b,k] = b9 + sum_{c,p} w9[c,p]*h3[b,c,idx[b,k,p]]
// ---------------------------------------------------------------------------
__global__ __launch_bounds__(256)
void softpool_gather(const unsigned short* __restrict__ raw3,
                     const float* __restrict__ ss3,
                     const int* __restrict__ idx,
                     const float* __restrict__ w9,
                     const float* __restrict__ b9,
                     float* __restrict__ glob, int N) {
  __shared__ float red[256];
  const int bk = blockIdx.x, b = bk >> 5, tid = threadIdx.x;
  float acc = 0.0f;
  for (int i = tid; i < 512; i += 256) {
    int c = i >> 4, p = i & 15;
    int n = idx[bk * 16 + p];
    float h = bf2f_bits(raw3[((size_t)b * 32 + c) * N + n]) * ss3[c] + ss3[32 + c];
    acc += w9[c * 16 + p] * h;
  }
  red[tid] = acc;
  __syncthreads();
  for (int s = 128; s > 0; s >>= 1) {
    if (tid < s) red[tid] += red[tid + s];
    __syncthreads();
  }
  if (tid == 0) glob[bk] = red[0] + b9[0];
}

// ---------------------------------------------------------------------------
// Host launcher
// ---------------------------------------------------------------------------
extern "C" void kernel_launch(void* const* d_in, const int* in_sizes, int n_in,
                              void* d_out, int out_size, void* d_ws, size_t ws_size,
                              hipStream_t stream) {
  constexpr int B = 16, N = 16384;
  const float* x   = (const float*)d_in[0];
  const float* w1  = (const float*)d_in[1];  const float* b1  = (const float*)d_in[2];
  const float* g1  = (const float*)d_in[3];  const float* be1 = (const float*)d_in[4];
  const float* w2  = (const float*)d_in[5];  const float* b2  = (const float*)d_in[6];
  const float* g2  = (const float*)d_in[7];  const float* be2 = (const float*)d_in[8];
  const float* w3  = (const float*)d_in[9];  const float* b3  = (const float*)d_in[10];
  const float* g3  = (const float*)d_in[11]; const float* be3 = (const float*)d_in[12];
  const float* w9  = (const float*)d_in[13]; const float* b9  = (const float*)d_in[14];
  const float* w4  = (const float*)d_in[15]; const float* b4  = (const float*)d_in[16];
  const float* g4  = (const float*)d_in[17]; const float* be4 = (const float*)d_in[18];
  const float* w5  = (const float*)d_in[19]; const float* b5  = (const float*)d_in[20];
  const float* g5  = (const float*)d_in[21]; const float* be5 = (const float*)d_in[22];
  const float* w6  = (const float*)d_in[23]; const float* b6  = (const float*)d_in[24];
  const float* g6  = (const float*)d_in[25]; const float* be6 = (const float*)d_in[26];
  const float* w7  = (const float*)d_in[27]; const float* b7  = (const float*)d_in[28];

  // ---- workspace layout (assumes ws_size >= ~512MB; raw6 reuses raw2) ----
  char* ws = (char*)d_ws;
  size_t off = 0;
  auto take = [&](size_t bytes) -> char* {
    char* p = ws + off; off = (off + bytes + 255) & ~(size_t)255; return p;
  };
  float* stats = (float*)take(2240 * sizeof(float));  // 2*(64+128+32+512+256+128)
  float* ssbuf = (float*)take(2240 * sizeof(float));
  int*   idx   = (int*)take((size_t)B * 32 * 16 * sizeof(int));
  float* glob  = (float*)take((size_t)B * 32 * sizeof(float));
  unsigned short* raw0 = (unsigned short*)take((size_t)B * 4   * N * 2);
  unsigned short* raw1 = (unsigned short*)take((size_t)B * 64  * N * 2);
  unsigned short* raw2 = (unsigned short*)take((size_t)B * 128 * N * 2);
  unsigned short* raw3 = (unsigned short*)take((size_t)B * 32  * N * 2);
  unsigned short* raw4 = (unsigned short*)take((size_t)B * 512 * N * 2);
  unsigned short* raw5 = (unsigned short*)take((size_t)B * 256 * N * 2);
  unsigned short* raw6 = raw2;   // raw2 dead after conv3; conv6 runs later

  float *s1 = stats,       *s2 = stats + 128, *s3 = stats + 384,
        *s4 = stats + 448, *s5 = stats + 1472, *s6 = stats + 1984;
  float *q1 = ssbuf,       *q2 = ssbuf + 128, *q3 = ssbuf + 384,
        *q4 = ssbuf + 448, *q5 = ssbuf + 1472, *q6 = ssbuf + 1984;

  hipMemsetAsync(stats, 0, 2240 * sizeof(float), stream);

  const float inv = 1.0f / ((float)B * (float)N);
  auto smem = [](int Cin) {
    int KP = (Cin + 31) & ~31;
    return (size_t)(192 * (KP + 8) * 2);  // (64 W rows + 128 X rows) * LDSTR * 2B
  };
  dim3 blk(256);

  // x -> bf16
  {
    int tot = B * 4 * N;
    to_bf16_kernel<<<(tot + 255) / 256, 256, 0, stream>>>(x, raw0, tot);
  }
  // conv1: 4 -> 64 (no input BN), emit stats1
  conv1x1_wmma<0, false, false, true, false>
      <<<dim3(N / 128, 1, B), blk, smem(4), stream>>>(
          raw0, nullptr, nullptr, w1, b1, raw1, nullptr, s1, 4, 64, N);
  bn_finalize<<<1, 64, 0, stream>>>(s1, g1, be1, q1, 64, inv);
  // conv2: BN1+ReLU fused on load, 64 -> 128
  conv1x1_wmma<0, true, true, true, false>
      <<<dim3(N / 128, 2, B), blk, smem(64), stream>>>(
          raw1, nullptr, q1, w2, b2, raw2, nullptr, s2, 64, 128, N);
  bn_finalize<<<1, 128, 0, stream>>>(s2, g2, be2, q2, 128, inv);
  // conv3: BN2+ReLU fused, 128 -> 32
  conv1x1_wmma<0, true, true, true, false>
      <<<dim3(N / 128, 1, B), blk, smem(128), stream>>>(
          raw2, nullptr, q2, w3, b3, raw3, nullptr, s3, 128, 32, N);
  bn_finalize<<<1, 32, 0, stream>>>(s3, g3, be3, q3, 32, inv);
  // SoftPool + conv9
  softpool_select<<<B * 32, 256, 0, stream>>>(raw3, q3, idx, N);
  softpool_gather<<<B * 32, 256, 0, stream>>>(raw3, q3, idx, w9, b9, glob, N);
  // conv4: [glob(32) ++ BN1+ReLU(h1)(64)] -> 512
  conv1x1_wmma<32, true, true, true, false>
      <<<dim3(N / 128, 8, B), blk, smem(96), stream>>>(
          raw1, glob, q1, w4, b4, raw4, nullptr, s4, 96, 512, N);
  bn_finalize<<<1, 512, 0, stream>>>(s4, g4, be4, q4, 512, inv);
  // conv5: 512 -> 256
  conv1x1_wmma<0, true, true, true, false>
      <<<dim3(N / 128, 4, B), blk, smem(512), stream>>>(
          raw4, nullptr, q4, w5, b5, raw5, nullptr, s5, 512, 256, N);
  bn_finalize<<<1, 256, 0, stream>>>(s5, g5, be5, q5, 256, inv);
  // conv6: 256 -> 128
  conv1x1_wmma<0, true, true, true, false>
      <<<dim3(N / 128, 2, B), blk, smem(256), stream>>>(
          raw5, nullptr, q5, w6, b6, raw6, nullptr, s6, 256, 128, N);
  bn_finalize<<<1, 128, 0, stream>>>(s6, g6, be6, q6, 128, inv);
  // conv7 head: 128 -> 3, tanh, fp32 out
  conv1x1_wmma<0, true, true, false, true>
      <<<dim3(N / 128, 1, B), blk, smem(128), stream>>>(
          raw6, nullptr, q6, w7, b7, nullptr, (float*)d_out, nullptr, 128, 3, N);
}
